// MambaEncoder_73624329388591
// MI455X (gfx1250) — matrix operand order repaired and checked by reading, not compiled, not run on hardware
//
#include <hip/hip_runtime.h>
#include <math.h>

typedef __attribute__((ext_vector_type(16))) __bf16 v16bf;
typedef __attribute__((ext_vector_type(8)))  float  v8f;
typedef __attribute__((ext_vector_type(4)))  float  f4;

__device__ __forceinline__ float sigmoidf_(float x){ return 1.f/(1.f+__expf(-x)); }

// ============================================================================
// WMMA GEMM:  C[M,N] = A[M,K] * B[K,N] (+ bias[M])
// One wave computes a 16x64 tile (4 accumulators along N), K-loop step 32 bf16.
// A (16x32 bf16), contiguous k order: lanes 0-15 = M rows; half0 ks {0-15},
// half1 ks {16-31}  ->  one contiguous 16-float run -> 4x b128 loads.
// B (32x16): lanes 0-15 hold K=0..15, lanes 16-31 hold K=16..31 (coalesced).
// B fetched via a single running cursor + immediate offsets {0,64,128,192}B
// (interior tiles); edge tiles take a clamped-cursor path (wave-uniform branch).
// Only the K tail needs zero-fill (single branchless remainder step).
// ============================================================================
__global__ __launch_bounds__(32)
void wmma_gemm_k(const float* __restrict__ A, const float* __restrict__ B,
                 const float* __restrict__ bias, float* __restrict__ C,
                 int M, int N, int K)
{
  const int lane = threadIdx.x & 31;
  const int half = lane >> 4;
  const int lid  = lane & 15;

  const int m   = blockIdx.x * 16 + lid;
  const int m_c = (m < M) ? m : (M - 1);
  const float* __restrict__ Arow = A + (size_t)m_c * K;
  const float* aptr = Arow + half * 16;

  const int n0 = blockIdx.y * 64;
  v8f acc0 = {}, acc1 = {}, acc2 = {}, acc3 = {};
  const int K32 = K & ~31;
  const size_t stepN  = (size_t)N;
  const size_t skip16 = (size_t)16 * N;

  if (n0 + 64 <= N) {
    // ---------------- interior tile: single cursor, immediate offsets ----
    const float* bp = B + (size_t)(half * 16) * N + (n0 + lid);
    for (int k0 = 0; k0 < K32; k0 += 32) {
      f4 a0 = *(const f4*)(aptr);
      f4 a1 = *(const f4*)(aptr + 4);
      f4 a2 = *(const f4*)(aptr + 8);
      f4 a3 = *(const f4*)(aptr + 12);
      aptr += 32;
      if (k0 + 32 < K32) __builtin_prefetch(aptr, 0, 1);

      v16bf av;
      av[0]=(__bf16)a0.x;  av[1]=(__bf16)a0.y;  av[2]=(__bf16)a0.z;  av[3]=(__bf16)a0.w;
      av[4]=(__bf16)a1.x;  av[5]=(__bf16)a1.y;  av[6]=(__bf16)a1.z;  av[7]=(__bf16)a1.w;
      av[8]=(__bf16)a2.x;  av[9]=(__bf16)a2.y;  av[10]=(__bf16)a2.z; av[11]=(__bf16)a2.w;
      av[12]=(__bf16)a3.x; av[13]=(__bf16)a3.y; av[14]=(__bf16)a3.z; av[15]=(__bf16)a3.w;

      v16bf bv0, bv1, bv2, bv3;
#pragma unroll
      for (int i = 0; i < 16; ++i) {
        bv0[i] = (__bf16)bp[0];
        bv1[i] = (__bf16)bp[16];
        bv2[i] = (__bf16)bp[32];
        bv3[i] = (__bf16)bp[48];
        bp += stepN;
      }
      bp += skip16;
      acc0 = __builtin_amdgcn_wmma_f32_16x16x32_bf16(false, av, false, bv0, (short)0, acc0, false, false);
      acc1 = __builtin_amdgcn_wmma_f32_16x16x32_bf16(false, av, false, bv1, (short)0, acc1, false, false);
      acc2 = __builtin_amdgcn_wmma_f32_16x16x32_bf16(false, av, false, bv2, (short)0, acc2, false, false);
      acc3 = __builtin_amdgcn_wmma_f32_16x16x32_bf16(false, av, false, bv3, (short)0, acc3, false, false);
    }
    if (K32 < K) {
      v16bf av, bv0, bv1, bv2, bv3;
#pragma unroll
      for (int i = 0; i < 16; ++i) {
        int ka = K32 + half * 16 + i;
        float a = Arow[(ka < K) ? ka : (K - 1)];
        if (ka >= K) a = 0.f;
        av[i] = (__bf16)a;
        int kb = K32 + half * 16 + i;
        const float* q = B + (size_t)((kb < K) ? kb : (K - 1)) * N + (n0 + lid);
        float b0 = q[0], b1 = q[16], b2 = q[32], b3 = q[48];
        if (kb >= K) { b0 = 0.f; b1 = 0.f; b2 = 0.f; b3 = 0.f; }
        bv0[i] = (__bf16)b0; bv1[i] = (__bf16)b1; bv2[i] = (__bf16)b2; bv3[i] = (__bf16)b3;
      }
      acc0 = __builtin_amdgcn_wmma_f32_16x16x32_bf16(false, av, false, bv0, (short)0, acc0, false, false);
      acc1 = __builtin_amdgcn_wmma_f32_16x16x32_bf16(false, av, false, bv1, (short)0, acc1, false, false);
      acc2 = __builtin_amdgcn_wmma_f32_16x16x32_bf16(false, av, false, bv2, (short)0, acc2, false, false);
      acc3 = __builtin_amdgcn_wmma_f32_16x16x32_bf16(false, av, false, bv3, (short)0, acc3, false, false);
    }
  } else {
    // ---------------- edge tile: 4 clamped cursors --------------------------
    int nc0 = n0 +  0 + lid; if (nc0 >= N) nc0 = N - 1;
    int nc1 = n0 + 16 + lid; if (nc1 >= N) nc1 = N - 1;
    int nc2 = n0 + 32 + lid; if (nc2 >= N) nc2 = N - 1;
    int nc3 = n0 + 48 + lid; if (nc3 >= N) nc3 = N - 1;
    const float* bp0 = B + (size_t)(half * 16) * N + nc0;
    const float* bp1 = B + (size_t)(half * 16) * N + nc1;
    const float* bp2 = B + (size_t)(half * 16) * N + nc2;
    const float* bp3 = B + (size_t)(half * 16) * N + nc3;
    for (int k0 = 0; k0 < K32; k0 += 32) {
      f4 a0 = *(const f4*)(aptr);
      f4 a1 = *(const f4*)(aptr + 4);
      f4 a2 = *(const f4*)(aptr + 8);
      f4 a3 = *(const f4*)(aptr + 12);
      aptr += 32;

      v16bf av;
      av[0]=(__bf16)a0.x;  av[1]=(__bf16)a0.y;  av[2]=(__bf16)a0.z;  av[3]=(__bf16)a0.w;
      av[4]=(__bf16)a1.x;  av[5]=(__bf16)a1.y;  av[6]=(__bf16)a1.z;  av[7]=(__bf16)a1.w;
      av[8]=(__bf16)a2.x;  av[9]=(__bf16)a2.y;  av[10]=(__bf16)a2.z; av[11]=(__bf16)a2.w;
      av[12]=(__bf16)a3.x; av[13]=(__bf16)a3.y; av[14]=(__bf16)a3.z; av[15]=(__bf16)a3.w;

      v16bf bv0, bv1, bv2, bv3;
#pragma unroll
      for (int i = 0; i < 16; ++i) {
        bv0[i] = (__bf16)bp0[0]; bp0 += stepN;
        bv1[i] = (__bf16)bp1[0]; bp1 += stepN;
        bv2[i] = (__bf16)bp2[0]; bp2 += stepN;
        bv3[i] = (__bf16)bp3[0]; bp3 += stepN;
      }
      bp0 += skip16; bp1 += skip16; bp2 += skip16; bp3 += skip16;
      acc0 = __builtin_amdgcn_wmma_f32_16x16x32_bf16(false, av, false, bv0, (short)0, acc0, false, false);
      acc1 = __builtin_amdgcn_wmma_f32_16x16x32_bf16(false, av, false, bv1, (short)0, acc1, false, false);
      acc2 = __builtin_amdgcn_wmma_f32_16x16x32_bf16(false, av, false, bv2, (short)0, acc2, false, false);
      acc3 = __builtin_amdgcn_wmma_f32_16x16x32_bf16(false, av, false, bv3, (short)0, acc3, false, false);
    }
    if (K32 < K) {
      v16bf av, bv0, bv1, bv2, bv3;
#pragma unroll
      for (int i = 0; i < 16; ++i) {
        int ka = K32 + half * 16 + i;
        float a = Arow[(ka < K) ? ka : (K - 1)];
        if (ka >= K) a = 0.f;
        av[i] = (__bf16)a;
        int kb = K32 + half * 16 + i;
        size_t ro = (size_t)((kb < K) ? kb : (K - 1)) * N;
        float b0 = B[ro + nc0], b1 = B[ro + nc1], b2 = B[ro + nc2], b3 = B[ro + nc3];
        if (kb >= K) { b0 = 0.f; b1 = 0.f; b2 = 0.f; b3 = 0.f; }
        bv0[i] = (__bf16)b0; bv1[i] = (__bf16)b1; bv2[i] = (__bf16)b2; bv3[i] = (__bf16)b3;
      }
      acc0 = __builtin_amdgcn_wmma_f32_16x16x32_bf16(false, av, false, bv0, (short)0, acc0, false, false);
      acc1 = __builtin_amdgcn_wmma_f32_16x16x32_bf16(false, av, false, bv1, (short)0, acc1, false, false);
      acc2 = __builtin_amdgcn_wmma_f32_16x16x32_bf16(false, av, false, bv2, (short)0, acc2, false, false);
      acc3 = __builtin_amdgcn_wmma_f32_16x16x32_bf16(false, av, false, bv3, (short)0, acc3, false, false);
    }
  }

  // ---------------- store ----------------
  const int row = blockIdx.x * 16 + half * 8;
#pragma unroll
  for (int v = 0; v < 8; ++v) {
    int r = row + v;
    if (r >= M) break;
    float bz = bias ? bias[r] : 0.f;
    float o0 = acc0[v] + bz, o1 = acc1[v] + bz, o2 = acc2[v] + bz, o3 = acc3[v] + bz;
    size_t rb = (size_t)r * N;
    int c0 = n0 + lid;
    if (c0      < N) C[rb + c0]      = o0;
    if (c0 + 16 < N) C[rb + c0 + 16] = o1;
    if (c0 + 32 < N) C[rb + c0 + 32] = o2;
    if (c0 + 48 < N) C[rb + c0 + 48] = o3;
  }
}

// ============================================================================
// Direct 3D convolution (NCDHW, OIDHW), one thread per output element.
// ============================================================================
__global__ void conv3d_k(const float* __restrict__ x, const float* __restrict__ w,
                         const float* __restrict__ bias, float* __restrict__ y,
                         int Cin, int Cout, int Di, int Hi, int Wi,
                         int Do, int Ho, int Wo, int ks, int stride, int pad)
{
  int idx = blockIdx.x * blockDim.x + threadIdx.x;
  int total = Cout * Do * Ho * Wo;
  if (idx >= total) return;
  int wo = idx % Wo; int t = idx / Wo;
  int ho = t % Ho; t /= Ho;
  int dz = t % Do; int co = t / Do;
  float s = bias[co];
  for (int ci = 0; ci < Cin; ++ci) {
    const float* xin = x + (size_t)ci * Di * Hi * Wi;
    const float* wk  = w + ((size_t)co * Cin + ci) * ks * ks * ks;
    for (int kz = 0; kz < ks; ++kz) {
      int iz = dz * stride - pad + kz; if (iz < 0 || iz >= Di) continue;
      for (int ky = 0; ky < ks; ++ky) {
        int iy = ho * stride - pad + ky; if (iy < 0 || iy >= Hi) continue;
        for (int kx = 0; kx < ks; ++kx) {
          int ix = wo * stride - pad + kx; if (ix < 0 || ix >= Wi) continue;
          s += wk[(kz * ks + ky) * ks + kx] * xin[((size_t)iz * Hi + iy) * Wi + ix];
        }
      }
    }
  }
  y[idx] = s;
}

// ============================================================================
// Instance norm over spatial dims; one block per channel. In-place safe.
// ============================================================================
__global__ void instnorm_k(const float* __restrict__ x, float* __restrict__ y,
                           int Len)
{
  int c = blockIdx.x;
  const float* xc = x + (size_t)c * Len;
  float* yc = y + (size_t)c * Len;
  __shared__ float ssum[256], ssq[256];
  float s = 0.f, q = 0.f;
  for (int i = threadIdx.x; i < Len; i += blockDim.x) {
    float v = xc[i]; s += v; q += v * v;
  }
  ssum[threadIdx.x] = s; ssq[threadIdx.x] = q;
  __syncthreads();
  for (int off = 128; off > 0; off >>= 1) {
    if ((int)threadIdx.x < off) {
      ssum[threadIdx.x] += ssum[threadIdx.x + off];
      ssq[threadIdx.x]  += ssq[threadIdx.x + off];
    }
    __syncthreads();
  }
  float mu  = ssum[0] / Len;
  float var = ssq[0] / Len - mu * mu;
  float inv = rsqrtf(var + 1e-5f);
  for (int i = threadIdx.x; i < Len; i += blockDim.x)
    yc[i] = (xc[i] - mu) * inv;
}

// Layer norm over channel dim for (C, L) channel-major data.
__global__ void layernorm_ch_k(const float* __restrict__ x, const float* __restrict__ g,
                               const float* __restrict__ b, float* __restrict__ y,
                               int C, int Len)
{
  int i = blockIdx.x * blockDim.x + threadIdx.x;
  if (i >= Len) return;
  float mu = 0.f;
  for (int c = 0; c < C; ++c) mu += x[(size_t)c * Len + i];
  mu /= C;
  float var = 0.f;
  for (int c = 0; c < C; ++c) { float d = x[(size_t)c * Len + i] - mu; var += d * d; }
  var /= C;
  float inv = rsqrtf(var + 1e-5f);
  for (int c = 0; c < C; ++c)
    y[(size_t)c * Len + i] = (x[(size_t)c * Len + i] - mu) * inv * g[c] + b[c];
}

// ============================================================================
// Elementwise kernels
// ============================================================================
__global__ void relu_k(float* x, int n){ int i=blockIdx.x*blockDim.x+threadIdx.x; if(i<n) x[i]=fmaxf(x[i],0.f); }
__global__ void add_k (float* d, const float* s, int n){ int i=blockIdx.x*blockDim.x+threadIdx.x; if(i<n) d[i]+=s[i]; }
__global__ void copy_k(float* d, const float* s, int n){ int i=blockIdx.x*blockDim.x+threadIdx.x; if(i<n) d[i]=s[i]; }
__global__ void gelu_k(float* x, int n){
  int i=blockIdx.x*blockDim.x+threadIdx.x; if(i>=n) return;
  float v=x[i]; x[i]=0.5f*v*(1.f+erff(v*0.70710678118654752f));
}
__global__ void softplus_k(float* x, int n){
  int i=blockIdx.x*blockDim.x+threadIdx.x; if(i>=n) return;
  float v=x[i]; x[i] = (v>20.f) ? v : log1pf(__expf(v));
}

// Reverse each row of (R, Len)
__global__ void reverse_rows_k(const float* __restrict__ x, float* __restrict__ y,
                               int R, int Len)
{
  int idx = blockIdx.x*blockDim.x+threadIdx.x;
  if (idx >= R*Len) return;
  int r = idx / Len, i = idx % Len;
  y[(size_t)r*Len + i] = x[(size_t)r*Len + (Len-1-i)];
}

// Per-row inner transpose: x viewed [R][P][Q] -> y [R][Q][P]; y[r][q][p]=x[r][p][q]
__global__ void row_transpose_k(const float* __restrict__ x, float* __restrict__ y,
                                int R, int P, int Q)
{
  int idx = blockIdx.x*blockDim.x+threadIdx.x;
  if (idx >= R*P*Q) return;
  int p = idx % P; int t = idx / P;
  int q = t % Q;  int r = t / Q;
  y[((size_t)r*Q + q)*P + p] = x[((size_t)r*P + p)*Q + q];
}

// Depthwise causal conv1d (k=4) + bias + SiLU
__global__ void dwconv4_silu_k(const float* __restrict__ x, const float* __restrict__ w,
                               const float* __restrict__ b, float* __restrict__ y,
                               int D, int Len)
{
  int idx = blockIdx.x*blockDim.x+threadIdx.x;
  if (idx >= D*Len) return;
  int d = idx / Len, i = idx % Len;
  float s = b[d];
#pragma unroll
  for (int j = 0; j < 4; ++j) {
    int t = i - 3 + j;
    if (t >= 0) s += w[d*4 + j] * x[(size_t)d*Len + t];
  }
  y[idx] = s * sigmoidf_(s);
}

// ============================================================================
// Selective scan: one thread per channel, 16 hidden states in registers.
// y = (scan + D*u) * silu(z)
// ============================================================================
__global__ void selective_scan_k(const float* __restrict__ u, const float* __restrict__ delta,
                                 const float* __restrict__ A_log, const float* __restrict__ Bm,
                                 const float* __restrict__ Cm, const float* __restrict__ Dp,
                                 const float* __restrict__ z, float* __restrict__ y,
                                 int D, int Len)
{
  int d = blockIdx.x*blockDim.x + threadIdx.x;
  if (d >= D) return;
  float A[16], h[16];
#pragma unroll
  for (int n = 0; n < 16; ++n) { A[n] = -__expf(A_log[(size_t)d*16 + n]); h[n] = 0.f; }
  float Dd = Dp[d];
  for (int t = 0; t < Len; ++t) {
    float dt = delta[(size_t)d*Len + t];
    float ut = u[(size_t)d*Len + t];
    float du = dt * ut;
    float acc = 0.f;
#pragma unroll
    for (int n = 0; n < 16; ++n) {
      h[n] = __expf(dt * A[n]) * h[n] + du * Bm[(size_t)n*Len + t];
      acc += h[n] * Cm[(size_t)n*Len + t];
    }
    float yy = acc + Dd * ut;
    float zt = z[(size_t)d*Len + t];
    y[(size_t)d*Len + t] = yy * (zt * sigmoidf_(zt));
  }
}

// ============================================================================
// ECA: channel pooling, 1D conv over channels, sigmoid gating
// ============================================================================
__global__ void pool_mean_k(const float* __restrict__ x, float* __restrict__ s, int Len)
{
  int c = blockIdx.x;
  __shared__ float ssum[256];
  float acc = 0.f;
  for (int i = threadIdx.x; i < Len; i += blockDim.x) acc += x[(size_t)c*Len + i];
  ssum[threadIdx.x] = acc;
  __syncthreads();
  for (int off = 128; off > 0; off >>= 1) {
    if ((int)threadIdx.x < off) ssum[threadIdx.x] += ssum[threadIdx.x + off];
    __syncthreads();
  }
  if (threadIdx.x == 0) s[c] = ssum[0] / Len;
}

__global__ void eca_conv_k(const float* __restrict__ s, const float* __restrict__ w,
                           float* __restrict__ t, int C, int k)
{
  int c = blockIdx.x*blockDim.x + threadIdx.x;
  if (c >= C) return;
  int pad = (k - 1) / 2;
  float acc = 0.f;
  for (int j = 0; j < k; ++j) {
    int cc = c + j - pad;
    if (cc >= 0 && cc < C) acc += w[j] * s[cc];
  }
  t[c] = acc;
}

__global__ void chscale_sig_k(const float* __restrict__ x, const float* __restrict__ t,
                              float* __restrict__ y, int C, int Len)
{
  int idx = blockIdx.x*blockDim.x + threadIdx.x;
  if (idx >= C*Len) return;
  int c = idx / Len;
  y[idx] = x[idx] * sigmoidf_(t[c]);
}

// ============================================================================
// Host orchestration
// ============================================================================
static inline dim3 ewg(int n){ return dim3((n + 255) / 256); }

struct DirP { const float *conv_w, *conv_b, *x_proj, *dt_w, *dt_b, *A_log, *D; };
struct BlkP { const float *ln_g, *ln_b, *in_proj, *out_proj, *eca_w; DirP dir[3]; };
struct StageP {
  const float* g[8];           // p1w p1b p2w p2b p3w p3b p4w p4b
  BlkP blk[2];
  const float *fc1w, *fc1b, *fc2w, *fc2b;
};

extern "C" void kernel_launch(void* const* d_in, const int* in_sizes, int n_in,
                              void* d_out, int out_size, void* d_ws, size_t ws_size,
                              hipStream_t stream)
{
  (void)in_sizes; (void)n_in; (void)out_size; (void)ws_size;

  static const int DIMS[4] = {48, 96, 192, 384};
  static const int SD[4] = {48, 24, 12, 6};
  static const int SH[4] = {24, 12, 6, 3};
  static const int SW[4] = {24, 12, 6, 3};
  static const int NS[4] = {48, 24, 12, 6};
  static const int DTR[4] = {3, 6, 12, 24};
  static const int ECAK[4] = {3, 3, 5, 5};

  // ---- decode inputs (setup_inputs dict order) ----
  int ii = 0;
  auto F = [&]() -> const float* { return (const float*)d_in[ii++]; };
  const float* x_in   = F();
  const float* stem_w = F();
  const float* stem_b = F();
  const float *ds_w[3], *ds_b[3];
  for (int i = 0; i < 3; ++i) { ds_w[i] = F(); ds_b[i] = F(); }
  StageP st[4];
  for (int s = 0; s < 4; ++s) {
    for (int j = 0; j < 8; ++j) st[s].g[j] = F();
    for (int b = 0; b < 2; ++b) {
      BlkP& B = st[s].blk[b];
      B.ln_g = F(); B.ln_b = F(); B.in_proj = F(); B.out_proj = F(); B.eca_w = F();
      for (int d = 0; d < 3; ++d) {
        DirP& Dp = B.dir[d];
        Dp.conv_w = F(); Dp.conv_b = F(); Dp.x_proj = F(); Dp.dt_w = F();
        Dp.dt_b = F(); Dp.A_log = F(); Dp.D = F();
      }
    }
    st[s].fc1w = F(); st[s].fc1b = F(); st[s].fc2w = F(); st[s].fc2b = F();
  }

  // ---- workspace bump allocator (sizes at stage-0 maxima, reused) ----
  char* wsb = (char*)d_ws;
  size_t off = 0;
  auto alloc = [&](size_t nf) -> float* {
    float* p = (float*)(wsb + off);
    off += ((nf * sizeof(float) + 255) / 256) * 256;
    return p;
  };
  const size_t CL  = (size_t)48 * 27648;   // C*L max
  const size_t EL  = (size_t)192 * 27648;  // e*L max (e = 4*C)
  const size_t DIL = (size_t)96 * 27648;   // di*L max
  const size_t XDL = (size_t)56 * 27648;   // (dt_rank+32)*L max

  float* h     = alloc(CL);
  float* hn    = alloc(CL);
  float* t1    = alloc(CL);
  float* t2    = alloc(CL);
  float* xz    = alloc(EL);
  float* xzd   = alloc(EL);
  float* xc    = alloc(DIL);
  float* xdbl  = alloc(XDL);
  float* delta = alloc(DIL);
  float* ydir  = alloc(DIL);
  float* ytmp  = alloc(DIL);
  float* yacc  = alloc(DIL);
  float* mlph  = alloc(DIL);   // 2C*L == di*L
  float* svec  = alloc(1024);
  float* svec2 = alloc(1024);

  float* out = (float*)d_out;
  size_t out_off = 0;

  auto gemm = [&](const float* A, const float* B, const float* bias, float* C,
                  int M, int N, int K) {
    dim3 grid((M + 15) / 16, (N + 63) / 64);
    wmma_gemm_k<<<grid, dim3(32), 0, stream>>>(A, B, bias, C, M, N, K);
  };

  for (int s = 0; s < 4; ++s) {
    const int C = DIMS[s];
    const int D = SD[s], H = SH[s], W = SW[s];
    const int L = D * H * W;
    const int n = C * L;
    const int DI = 2 * C;                // d_inner
    const int E  = 4 * C;                // 2*d_inner
    const int dtr = DTR[s];
    const int ns = NS[s];
    const int cs = L / ns;

    // ---- stem / downsample ----
    if (s == 0) {
      int tot = 48 * 48 * 24 * 24;
      conv3d_k<<<ewg(tot), dim3(256), 0, stream>>>(
          x_in, stem_w, stem_b, h, 1, 48, 96, 48, 48, 48, 24, 24, 7, 2, 3);
    } else {
      // hn holds instance_norm of previous h from end of prior stage
      int tot = C * L;
      conv3d_k<<<ewg(tot), dim3(256), 0, stream>>>(
          hn, ds_w[s - 1], ds_b[s - 1], h,
          DIMS[s - 1], C, SD[s - 1], SH[s - 1], SW[s - 1], D, H, W, 2, 2, 0);
    }

    // ---- GSC block ----
    {
      conv3d_k<<<ewg(n), dim3(256), 0, stream>>>(h, st[s].g[0], st[s].g[1], t1,
          C, C, D, H, W, D, H, W, 3, 1, 1);
      instnorm_k<<<dim3(C), dim3(256), 0, stream>>>(t1, t1, L);
      relu_k<<<ewg(n), dim3(256), 0, stream>>>(t1, n);

      conv3d_k<<<ewg(n), dim3(256), 0, stream>>>(t1, st[s].g[2], st[s].g[3], t2,
          C, C, D, H, W, D, H, W, 3, 1, 1);
      instnorm_k<<<dim3(C), dim3(256), 0, stream>>>(t2, t2, L);
      relu_k<<<ewg(n), dim3(256), 0, stream>>>(t2, n);            // x1 in t2

      conv3d_k<<<ewg(n), dim3(256), 0, stream>>>(h, st[s].g[4], st[s].g[5], t1,
          C, C, D, H, W, D, H, W, 1, 1, 0);
      instnorm_k<<<dim3(C), dim3(256), 0, stream>>>(t1, t1, L);
      relu_k<<<ewg(n), dim3(256), 0, stream>>>(t1, n);            // x2 in t1

      add_k<<<ewg(n), dim3(256), 0, stream>>>(t1, t2, n);         // x1 + x2
      conv3d_k<<<ewg(n), dim3(256), 0, stream>>>(t1, st[s].g[6], st[s].g[7], t2,
          C, C, D, H, W, D, H, W, 1, 1, 0);
      instnorm_k<<<dim3(C), dim3(256), 0, stream>>>(t2, t2, L);
      relu_k<<<ewg(n), dim3(256), 0, stream>>>(t2, n);
      add_k<<<ewg(n), dim3(256), 0, stream>>>(h, t2, n);          // h = y + x
    }

    // ---- Mamba blocks ----
    for (int b = 0; b < 2; ++b) {
      const BlkP& B = st[s].blk[b];

      layernorm_ch_k<<<ewg(L), dim3(256), 0, stream>>>(h, B.ln_g, B.ln_b, hn, C, L);
      gemm(B.in_proj, hn, nullptr, xz, E, L, C);                  // xz (E, L)

      for (int d = 0; d < 3; ++d) {
        const DirP& P = B.dir[d];
        const float* xin = xz;
        if (d == 1) {        // backward: reverse along L
          reverse_rows_k<<<ewg(E * L), dim3(256), 0, stream>>>(xz, xzd, E, L);
          xin = xzd;
        } else if (d == 2) { // slice direction: inner transpose (ns, cs)
          row_transpose_k<<<ewg(E * L), dim3(256), 0, stream>>>(xz, xzd, E, ns, cs);
          xin = xzd;
        }
        const float* xb = xin;
        const float* zb = xin + (size_t)DI * L;

        dwconv4_silu_k<<<ewg(DI * L), dim3(256), 0, stream>>>(
            xb, P.conv_w, P.conv_b, xc, DI, L);
        gemm(P.x_proj, xc, nullptr, xdbl, dtr + 32, L, DI);
        gemm(P.dt_w, xdbl, P.dt_b, delta, DI, L, dtr);
        softplus_k<<<ewg(DI * L), dim3(256), 0, stream>>>(delta, DI * L);

        const float* Bm = xdbl + (size_t)dtr * L;
        const float* Cm = xdbl + (size_t)(dtr + 16) * L;
        selective_scan_k<<<dim3((DI + 63) / 64), dim3(64), 0, stream>>>(
            xc, delta, P.A_log, Bm, Cm, P.D, zb, ydir, DI, L);

        if (d == 0) {
          copy_k<<<ewg(DI * L), dim3(256), 0, stream>>>(yacc, ydir, DI * L);
        } else if (d == 1) {
          reverse_rows_k<<<ewg(DI * L), dim3(256), 0, stream>>>(ydir, ytmp, DI, L);
          add_k<<<ewg(DI * L), dim3(256), 0, stream>>>(yacc, ytmp, DI * L);
        } else {
          row_transpose_k<<<ewg(DI * L), dim3(256), 0, stream>>>(ydir, ytmp, DI, cs, ns);
          add_k<<<ewg(DI * L), dim3(256), 0, stream>>>(yacc, ytmp, DI * L);
        }
      }

      gemm(B.out_proj, yacc, nullptr, t1, C, L, DI);              // (C, L)
      add_k<<<ewg(n), dim3(256), 0, stream>>>(t1, h, n);          // residual

      pool_mean_k<<<dim3(C), dim3(256), 0, stream>>>(t1, svec, L);
      eca_conv_k<<<dim3((C + 255) / 256), dim3(256), 0, stream>>>(
          svec, B.eca_w, svec2, C, ECAK[s]);
      chscale_sig_k<<<ewg(n), dim3(256), 0, stream>>>(t1, svec2, h, C, L);
    }

    // ---- stage head: instance_norm + channel MLP ----
    instnorm_k<<<dim3(C), dim3(256), 0, stream>>>(h, hn, L);      // hn reused next stage
    gemm(st[s].fc1w, hn, st[s].fc1b, mlph, 2 * C, L, C);
    gelu_k<<<ewg(2 * C * L), dim3(256), 0, stream>>>(mlph, 2 * C * L);
    gemm(st[s].fc2w, mlph, st[s].fc2b, out + out_off, C, L, 2 * C);
    out_off += (size_t)C * L;
  }
}
